// TemporalGraphGATGRU_22840636080823
// MI455X (gfx1250) — matrix operand order repaired and checked by reading, not compiled, not run on hardware
//
#include <hip/hip_runtime.h>
#include <hip/hip_bf16.h>
#include <math.h>

// ---------------------------------------------------------------------------
// TemporalGraphGATGRU for MI455X (gfx1250, wave32, WMMA)
// Dense GEMMs (GAT1 x@W1, GAT2 h@W2, decoder MLP) run on
// v_wmma_f32_16x16x32_f16 with f32 accumulation. Weights are pre-packed into
// a WMMA B-operand register layout (k = 16*half + j) so every operand load is a
// contiguous global_load_b128 pair; each wave computes a 16x64 strip (4 WMMAs
// per K-slab sharing one A operand).
// ---------------------------------------------------------------------------

#define HEADS 4
#define NEG_SLOPE 0.2f

typedef __attribute__((ext_vector_type(16))) _Float16 v16h;
typedef __attribute__((ext_vector_type(8)))  float    v8f;

// ---------------- small utility kernels ----------------

__global__ void fill_f32(float* p, float v, int n) {
  int i = blockIdx.x * blockDim.x + threadIdx.x;
  if (i < n) p[i] = v;
}

__global__ void copy_f32(float* d, const float* s, int n) {
  int i = blockIdx.x * blockDim.x + threadIdx.x;
  if (i < n) d[i] = s[i];
}

__global__ void f32_to_f16_k(const float* s, _Float16* d, int n) {
  int i = blockIdx.x * blockDim.x + threadIdx.x;
  if (i < n) d[i] = (_Float16)s[i];
}

__device__ __forceinline__ void atomicMaxFloat(float* addr, float val) {
  // int/uint ordering trick: correct for all sign combinations.
  if (val >= 0.0f) atomicMax((int*)addr, __float_as_int(val));
  else             atomicMin((unsigned int*)addr, __float_as_uint(val));
}

// ---------------- weight packing into WMMA B-operand layout ----------------
// B 32x16 f16 (wave32), contiguous k order used here: lane L holds column n=L&15 of tile
// nt, half=L>>4, element j <-> K = kt*32 + half*16 + j.
// Packed layout: Bp[(((kt*tilesN)+nt)*32 + lane)*16 + j]  -> each lane's
// 16 halves are one contiguous 32-byte block (two global_load_b128).
__global__ void pack_b_wmma(const float* __restrict__ B, _Float16* __restrict__ Bp,
                            int K, int Nn) {
  int idx = blockIdx.x * blockDim.x + threadIdx.x;
  if (idx >= K * Nn) return;
  int k = idx / Nn, n = idx % Nn;
  int kt = k >> 5, kk = k & 31, half = kk >> 4, j = kk & 15;
  int nt = n >> 4;
  int lane = half * 16 + (n & 15);
  size_t dst = (((size_t)kt * (Nn >> 4) + nt) * 32 + lane) * 16 + j;
  Bp[dst] = (_Float16)B[idx];
}

// ---------------- WMMA GEMM: C[M,N] = A[M,K]*Bpacked (+bias)(+relu) --------
// A row-major f16; B pre-packed (pack_b_wmma); C row-major f32.
// One wave per 16(M) x 64(N) strip: 4 accumulators share one A operand.
// A 16x32 f16 layout: lane L (m=L&15, half=L>>4): elem j -> K=half*16+j
//                     (one contiguous 32B chunk, matching packed B).
// C/D 16x16 f32: VGPR r -> M=r+8*half, N=lane&15.
__global__ __launch_bounds__(256)
void gemm_f16_wmma(const _Float16* __restrict__ A, const _Float16* __restrict__ Bp,
                   float* __restrict__ C, const float* __restrict__ bias,
                   int M, int Nn, int K, int relu) {
  int wave = (blockIdx.x * blockDim.x + threadIdx.x) >> 5;
  int lane = threadIdx.x & 31;
  int tilesN  = Nn >> 4;
  int groupsN = Nn >> 6;                 // 4 tiles of 16 per wave
  int tm = wave / groupsN;
  int tg = wave % groupsN;
  if (tm * 16 >= M) return;              // wave-uniform: EXEC stays all-1s
  int half = lane >> 4;
  int mn = lane & 15;

  v8f acc0 = {}, acc1 = {}, acc2 = {}, acc3 = {};
  const _Float16* Ap0 = A + (size_t)(tm * 16 + mn) * K;
  for (int k0 = 0; k0 < K; k0 += 32) {
    v16h a;
    const _Float16* Ap = Ap0 + k0 + half * 16;
    __builtin_prefetch(Ap0 + k0 + 32, 0, 1);   // global_prefetch_b8 next slab
#pragma unroll
    for (int j = 0; j < 8; ++j) { a[j] = Ap[j]; a[j + 8] = Ap[8 + j]; }

    const _Float16* Bb =
        Bp + (((size_t)(k0 >> 5) * tilesN + tg * 4) * 32 + lane) * 16;
    v16h b0 = *(const v16h*)(Bb);
    v16h b1 = *(const v16h*)(Bb + 32 * 16);
    v16h b2 = *(const v16h*)(Bb + 64 * 16);
    v16h b3 = *(const v16h*)(Bb + 96 * 16);
    acc0 = __builtin_amdgcn_wmma_f32_16x16x32_f16(false, a, false, b0, (short)0, acc0, false, false);
    acc1 = __builtin_amdgcn_wmma_f32_16x16x32_f16(false, a, false, b1, (short)0, acc1, false, false);
    acc2 = __builtin_amdgcn_wmma_f32_16x16x32_f16(false, a, false, b2, (short)0, acc2, false, false);
    acc3 = __builtin_amdgcn_wmma_f32_16x16x32_f16(false, a, false, b3, (short)0, acc3, false, false);
  }

  v8f* accs[4] = { &acc0, &acc1, &acc2, &acc3 };
#pragma unroll
  for (int i = 0; i < 4; ++i) {
    int col = tg * 64 + i * 16 + mn;
    float bv = bias ? bias[col] : 0.0f;
    v8f av = *accs[i];
#pragma unroll
    for (int r = 0; r < 8; ++r) {
      float v = av[r] + bv;
      if (relu) v = fmaxf(v, 0.0f);
      C[(size_t)(tm * 16 + r + 8 * half) * Nn + col] = v;
    }
  }
}

// ---------------- GAT node attention scores ----------------
// as_n[n,h] = sum_c xs[n,h*C+c]*a_s[h,c];  ad_n likewise.
__global__ void node_scores(const float* __restrict__ xs, const float* __restrict__ a_s,
                            const float* __restrict__ a_d, float* as_n, float* ad_n,
                            int n, int C) {
  int idx = blockIdx.x * blockDim.x + threadIdx.x;
  if (idx >= n * HEADS) return;
  int node = idx / HEADS, h = idx % HEADS;
  const float* xp = xs + (size_t)node * HEADS * C + h * C;
  float s = 0.f, d = 0.f;
  for (int c = 0; c < C; ++c) { s += xp[c] * a_s[h * C + c]; d += xp[c] * a_d[h * C + c]; }
  as_n[idx] = s;
  ad_n[idx] = d;
}

// ---------------- edge softmax passes (E' = E + N self loops) ----------------
__global__ void edge_pass1(const int* __restrict__ srcp, const int* __restrict__ dstp,
                           const float* __restrict__ as_n, const float* __restrict__ ad_n,
                           float* e_buf, float* m_buf, int EP, int Eraw) {
  int idx = blockIdx.x * blockDim.x + threadIdx.x;
  if (idx >= EP * HEADS) return;
  int e = idx / HEADS, h = idx % HEADS;
  int s = (e < Eraw) ? srcp[e] : (e - Eraw);
  int d = (e < Eraw) ? dstp[e] : (e - Eraw);
  float v = as_n[s * HEADS + h] + ad_n[d * HEADS + h];
  v = (v > 0.f) ? v : NEG_SLOPE * v;          // leaky_relu
  e_buf[idx] = v;
  atomicMaxFloat(&m_buf[d * HEADS + h], v);
}

__global__ void edge_pass2(const int* __restrict__ srcp, const int* __restrict__ dstp,
                           float* e_buf, const float* __restrict__ m_buf,
                           float* d_buf, int EP, int Eraw) {
  int idx = blockIdx.x * blockDim.x + threadIdx.x;
  if (idx >= EP * HEADS) return;
  int e = idx / HEADS, h = idx % HEADS;
  int d = (e < Eraw) ? dstp[e] : (e - Eraw);
  float ex = __expf(e_buf[idx] - m_buf[d * HEADS + h]);
  e_buf[idx] = ex;
  atomicAdd(&d_buf[d * HEADS + h], ex);
}

// layer 1: concat heads; xs layout [N, HEADS*64], channel c -> head c>>6
__global__ void edge_pass3_concat(const int* __restrict__ srcp, const int* __restrict__ dstp,
                                  const float* __restrict__ e_buf, const float* __restrict__ d_buf,
                                  const float* __restrict__ xs, float* gout,
                                  int EP, int Eraw, int H) {
  int idx = blockIdx.x * blockDim.x + threadIdx.x;
  if (idx >= EP * H) return;
  int e = idx / H, c = idx % H;
  int h = c >> 6;                                    // 64 ch per head
  int s = (e < Eraw) ? srcp[e] : (e - Eraw);
  int d = (e < Eraw) ? dstp[e] : (e - Eraw);
  float alpha = e_buf[e * HEADS + h] / (d_buf[d * HEADS + h] + 1e-16f);
  atomicAdd(&gout[(size_t)d * H + c], xs[(size_t)s * H + c] * alpha);
}

// layer 2: mean over heads; xs layout [N, HEADS*256]
__global__ void edge_pass3_mean(const int* __restrict__ srcp, const int* __restrict__ dstp,
                                const float* __restrict__ e_buf, const float* __restrict__ d_buf,
                                const float* __restrict__ xs, float* gout,
                                int EP, int Eraw, int H) {
  int idx = blockIdx.x * blockDim.x + threadIdx.x;
  if (idx >= EP * H) return;
  int e = idx / H, c = idx % H;
  int s = (e < Eraw) ? srcp[e] : (e - Eraw);
  int d = (e < Eraw) ? dstp[e] : (e - Eraw);
  float acc = 0.f;
#pragma unroll
  for (int h = 0; h < HEADS; ++h) {
    float alpha = e_buf[e * HEADS + h] / (d_buf[d * HEADS + h] + 1e-16f);
    acc += alpha * xs[(size_t)s * HEADS * H + h * H + c];
  }
  atomicAdd(&gout[(size_t)d * H + c], 0.25f * acc);
}

// ---------------- BatchNorm over node dim (one block per channel) ----------
__global__ void bn_kernel(const float* __restrict__ x, const float* __restrict__ b_pre,
                          const float* __restrict__ g, const float* __restrict__ be,
                          float* out, _Float16* out16, int n, int C, int do_relu) {
  int c = blockIdx.x;
  __shared__ float ssum[256], ssq[256];
  float s = 0.f, sq = 0.f;
  for (int i = threadIdx.x; i < n; i += blockDim.x) {
    float v = x[(size_t)i * C + c] + b_pre[c];
    s += v; sq += v * v;
  }
  ssum[threadIdx.x] = s; ssq[threadIdx.x] = sq;
  __syncthreads();
  for (int off = 128; off > 0; off >>= 1) {
    if (threadIdx.x < off) {
      ssum[threadIdx.x] += ssum[threadIdx.x + off];
      ssq[threadIdx.x]  += ssq[threadIdx.x + off];
    }
    __syncthreads();
  }
  float mu  = ssum[0] / n;
  float var = ssq[0] / n - mu * mu;
  float inv = rsqrtf(var + 1e-5f) * g[c];
  for (int i = threadIdx.x; i < n; i += blockDim.x) {
    float v = (x[(size_t)i * C + c] + b_pre[c] - mu) * inv + be[c];
    if (do_relu) v = fmaxf(v, 0.f);
    if (out)   out[(size_t)i * C + c] = v;
    if (out16) out16[(size_t)i * C + c] = (_Float16)v;
  }
}

// ---------------- graph mean pooling ----------------
__global__ void pool_kernel(const float* __restrict__ h2, const int* __restrict__ batch_idx,
                            float* emb_t, int n, int H, float inv_count) {
  int idx = blockIdx.x * blockDim.x + threadIdx.x;
  if (idx >= n * H) return;
  int node = idx / H, c = idx % H;
  atomicAdd(&emb_t[(size_t)batch_idx[node] * H + c], h2[idx] * inv_count);
}

// ---------------- GRU step: one block per batch row, H threads ----------------
__global__ void gru_step(const float* __restrict__ x, float* __restrict__ h,
                         const float* __restrict__ Wih, const float* __restrict__ Whh,
                         const float* __restrict__ bih, const float* __restrict__ bhh,
                         float* __restrict__ yout, int yrow_stride) {
  int b = blockIdx.x;
  int j = threadIdx.x;                 // H == 256 threads
  __shared__ float xr[256], hr[256];
  xr[j] = x[b * 256 + j];
  hr[j] = h[b * 256 + j];
  __syncthreads();
  float gi[3], gh[3];
#pragma unroll
  for (int g = 0; g < 3; ++g) {
    float si = bih[g * 256 + j], sh = bhh[g * 256 + j];
    const float* wi = Wih + (size_t)(g * 256 + j) * 256;
    const float* wh = Whh + (size_t)(g * 256 + j) * 256;
    for (int k = 0; k < 256; ++k) { si += xr[k] * wi[k]; sh += hr[k] * wh[k]; }
    gi[g] = si; gh[g] = sh;
  }
  float r  = 1.f / (1.f + __expf(-(gi[0] + gh[0])));
  float z  = 1.f / (1.f + __expf(-(gi[1] + gh[1])));
  float nn = tanhf(gi[2] + r * gh[2]);
  float hn = (1.f - z) * nn + z * hr[j];
  h[b * 256 + j] = hn;
  yout[b * yrow_stride + j] = hn;
}

// ---------------------------------------------------------------------------

extern "C" void kernel_launch(void* const* d_in, const int* in_sizes, int n_in,
                              void* d_out, int out_size, void* d_ws, size_t ws_size,
                              hipStream_t stream) {
  const int N = 8192, Bb = 16, T = 8, E = 32768, F = 128, H = 256, OUT = 128;
  const int EP = E + N;                         // edges + self loops

  const float* x_seq = (const float*)d_in[0];
  const float* h0    = (const float*)d_in[1];
  const float* W1    = (const float*)d_in[2];
  const float* a1s   = (const float*)d_in[3];
  const float* a1d   = (const float*)d_in[4];
  const float* b1    = (const float*)d_in[5];
  const float* W2    = (const float*)d_in[6];
  const float* a2s   = (const float*)d_in[7];
  const float* a2d   = (const float*)d_in[8];
  const float* b2    = (const float*)d_in[9];
  const float* g1    = (const float*)d_in[10];
  const float* be1   = (const float*)d_in[11];
  const float* g2    = (const float*)d_in[12];
  const float* be2   = (const float*)d_in[13];
  const float* Wih0  = (const float*)d_in[14];
  const float* Whh0  = (const float*)d_in[15];
  const float* bih0  = (const float*)d_in[16];
  const float* bhh0  = (const float*)d_in[17];
  const float* Wih1  = (const float*)d_in[18];
  const float* Whh1  = (const float*)d_in[19];
  const float* bih1  = (const float*)d_in[20];
  const float* bhh1  = (const float*)d_in[21];
  const float* Wd1   = (const float*)d_in[22];
  const float* bd1   = (const float*)d_in[23];
  const float* Wd2   = (const float*)d_in[24];
  const float* bd2   = (const float*)d_in[25];
  const int* edge_index = (const int*)d_in[26];
  const int* batch_idx  = (const int*)d_in[27];
  float* out = (float*)d_out;

  // ---- carve workspace (~64 MB) ----
  char* ws = (char*)d_ws;
  size_t off = 0;
  auto carve = [&](size_t bytes) -> char* {
    char* p = ws + off;
    off = (off + bytes + 255) & ~(size_t)255;
    return p;
  };
  _Float16* W1p   = (_Float16*)carve((size_t)F * H * 2);            // packed
  _Float16* W2p   = (_Float16*)carve((size_t)H * HEADS * H * 2);    // packed
  _Float16* Wd1p  = (_Float16*)carve((size_t)H * H * 2);            // packed
  _Float16* Wd2p  = (_Float16*)carve((size_t)H * OUT * 2);          // packed
  _Float16* x16   = (_Float16*)carve((size_t)N * F * 2);
  float*    xs1   = (float*)carve((size_t)N * H * 4);
  _Float16* h1_16 = (_Float16*)carve((size_t)N * H * 2);
  float*    xs2   = (float*)carve((size_t)N * HEADS * H * 4);
  float*    as_b  = (float*)carve((size_t)N * HEADS * 4);
  float*    ad_b  = (float*)carve((size_t)N * HEADS * 4);
  float*    m_b   = (float*)carve((size_t)N * HEADS * 4);
  float*    dn_b  = (float*)carve((size_t)N * HEADS * 4);
  float*    e_b   = (float*)carve((size_t)EP * HEADS * 4);
  float*    gout  = (float*)carve((size_t)N * H * 4);
  float*    h2f   = (float*)carve((size_t)N * H * 4);
  float*    emb   = (float*)carve((size_t)T * Bb * H * 4);
  float*    hL0   = (float*)carve((size_t)Bb * H * 4);
  float*    hL1   = (float*)carve((size_t)Bb * H * 4);
  float*    y0    = (float*)carve((size_t)T * Bb * H * 4);
  float*    y1    = (float*)carve((size_t)Bb * T * H * 4);
  _Float16* y116  = (_Float16*)carve((size_t)Bb * T * H * 2);
  float*    z1    = (float*)carve((size_t)Bb * T * H * 4);
  _Float16* z116  = (_Float16*)carve((size_t)Bb * T * H * 2);
  (void)ws_size;

  auto blk = [](int n) { return (n + 255) / 256; };
  const float NEG_INF = -__builtin_inff();

  // ---- one-time weight packing / init ----
  pack_b_wmma<<<blk(F * H), 256, 0, stream>>>(W1, W1p, F, H);
  pack_b_wmma<<<blk(H * HEADS * H), 256, 0, stream>>>(W2, W2p, H, HEADS * H);
  pack_b_wmma<<<blk(H * H), 256, 0, stream>>>(Wd1, Wd1p, H, H);
  pack_b_wmma<<<blk(H * OUT), 256, 0, stream>>>(Wd2, Wd2p, H, OUT);
  copy_f32<<<blk(Bb * H), 256, 0, stream>>>(hL0, h0, Bb * H);
  copy_f32<<<blk(Bb * H), 256, 0, stream>>>(hL1, h0 + Bb * H, Bb * H);
  fill_f32<<<blk(T * Bb * H), 256, 0, stream>>>(emb, 0.f, T * Bb * H);

  // ---- per-timestep GNN encoder ----
  for (int t = 0; t < T; ++t) {
    const float* xt   = x_seq + (size_t)t * N * F;
    const int*   srcp = edge_index + (size_t)t * 2 * E;
    const int*   dstp = srcp + E;

    // GAT layer 1: xs1 = x @ W1  (WMMA, 16x64 strip per wave)
    f32_to_f16_k<<<blk(N * F), 256, 0, stream>>>(xt, x16, N * F);
    {
      int waves = (N / 16) * (H / 64);
      gemm_f16_wmma<<<waves * 32 / 256, 256, 0, stream>>>(x16, W1p, xs1, nullptr, N, H, F, 0);
    }
    node_scores<<<blk(N * HEADS), 256, 0, stream>>>(xs1, a1s, a1d, as_b, ad_b, N, H / HEADS);
    fill_f32<<<blk(N * HEADS), 256, 0, stream>>>(m_b, NEG_INF, N * HEADS);
    fill_f32<<<blk(N * HEADS), 256, 0, stream>>>(dn_b, 0.f, N * HEADS);
    fill_f32<<<blk(N * H), 256, 0, stream>>>(gout, 0.f, N * H);
    edge_pass1<<<blk(EP * HEADS), 256, 0, stream>>>(srcp, dstp, as_b, ad_b, e_b, m_b, EP, E);
    edge_pass2<<<blk(EP * HEADS), 256, 0, stream>>>(srcp, dstp, e_b, m_b, dn_b, EP, E);
    edge_pass3_concat<<<blk(EP * H), 256, 0, stream>>>(srcp, dstp, e_b, dn_b, xs1, gout, EP, E, H);
    bn_kernel<<<H, 256, 0, stream>>>(gout, b1, g1, be1, nullptr, h1_16, N, H, 1);

    // GAT layer 2: xs2 = h1 @ W2  (WMMA, N=1024)
    {
      int waves = (N / 16) * (HEADS * H / 64);
      gemm_f16_wmma<<<waves * 32 / 256, 256, 0, stream>>>(h1_16, W2p, xs2, nullptr, N, HEADS * H, H, 0);
    }
    node_scores<<<blk(N * HEADS), 256, 0, stream>>>(xs2, a2s, a2d, as_b, ad_b, N, H);
    fill_f32<<<blk(N * HEADS), 256, 0, stream>>>(m_b, NEG_INF, N * HEADS);
    fill_f32<<<blk(N * HEADS), 256, 0, stream>>>(dn_b, 0.f, N * HEADS);
    fill_f32<<<blk(N * H), 256, 0, stream>>>(gout, 0.f, N * H);
    edge_pass1<<<blk(EP * HEADS), 256, 0, stream>>>(srcp, dstp, as_b, ad_b, e_b, m_b, EP, E);
    edge_pass2<<<blk(EP * HEADS), 256, 0, stream>>>(srcp, dstp, e_b, m_b, dn_b, EP, E);
    edge_pass3_mean<<<blk(EP * H), 256, 0, stream>>>(srcp, dstp, e_b, dn_b, xs2, gout, EP, E, H);
    bn_kernel<<<H, 256, 0, stream>>>(gout, b2, g2, be2, h2f, nullptr, N, H, 0);

    // mean pool per graph into emb[t]
    pool_kernel<<<blk(N * H), 256, 0, stream>>>(h2f, batch_idx, emb + (size_t)t * Bb * H,
                                                N, H, (float)Bb / (float)N);
  }

  // ---- GRU: 2 layers, sequential over T ----
  for (int t = 0; t < T; ++t)
    gru_step<<<Bb, H, 0, stream>>>(emb + (size_t)t * Bb * H, hL0, Wih0, Whh0, bih0, bhh0,
                                   y0 + (size_t)t * Bb * H, H);
  for (int t = 0; t < T; ++t)
    gru_step<<<Bb, H, 0, stream>>>(y0 + (size_t)t * Bb * H, hL1, Wih1, Whh1, bih1, bhh1,
                                   y1 + (size_t)t * H, T * H);   // batch-first [B,T,H]

  // ---- decoder MLP (WMMA) : out = relu(y1@Wd1+bd1)@Wd2+bd2 ----
  f32_to_f16_k<<<blk(Bb * T * H), 256, 0, stream>>>(y1, y116, Bb * T * H);
  {
    int waves = (Bb * T / 16) * (H / 64);                 // 128x256
    gemm_f16_wmma<<<waves * 32 / 256, 256, 0, stream>>>(y116, Wd1p, z1, bd1, Bb * T, H, H, 1);
  }
  f32_to_f16_k<<<blk(Bb * T * H), 256, 0, stream>>>(z1, z116, Bb * T * H);
  {
    int waves = (Bb * T / 16) * (OUT / 64);               // 128x128 -> d_out
    gemm_f16_wmma<<<waves * 32 / 256, 256, 0, stream>>>(z116, Wd2p, out, bd2, Bb * T, OUT, H, 0);
  }

  // ---- hn = stack([hL0, hL1]) ----
  copy_f32<<<blk(Bb * H), 256, 0, stream>>>(out + (size_t)Bb * T * OUT, hL0, Bb * H);
  copy_f32<<<blk(Bb * H), 256, 0, stream>>>(out + (size_t)Bb * T * OUT + Bb * H, hL1, Bb * H);
}
